// CodeTransformer_4810363372474
// MI455X (gfx1250) — matrix operand order repaired and checked by reading, not compiled, not run on hardware
//
#include <hip/hip_runtime.h>
#include <math.h>

// ---------------- problem constants ----------------
constexpr int Bb = 2, Ss = 1024, Dd = 1024, Hh = 16, Ll = 8, Ff = 4096, Vv = 32000, DHh = 64;

// ---------------- WMMA types ----------------
typedef __attribute__((ext_vector_type(16))) __bf16 v16bf;
typedef __attribute__((ext_vector_type(8)))  float  v8f;
typedef __attribute__((ext_vector_type(4)))  int    v4i;

union Frag { v16bf v; uint4 q[2]; };

#define AS1 __attribute__((address_space(1)))
#define AS3 __attribute__((address_space(3)))

// CDNA5 async global->LDS copy path (probe-gated; fallback = reg-prefetch pipeline)
#if __has_builtin(__builtin_amdgcn_global_load_async_to_lds_b128) && __has_builtin(__builtin_amdgcn_s_wait_asynccnt)
#define USE_ASYNC_LDS 1
// prototype (from hipcc diagnostic): (v4i __device__*, v4i __shared__*, imm offset, imm cpol)
#define ASYNC_CP16(g, l)                                                                 \
  __builtin_amdgcn_global_load_async_to_lds_b128((AS1 v4i*)(g), (AS3 v4i*)(l), 0, 0)
#else
#define USE_ASYNC_LDS 0
#endif

__device__ __forceinline__ unsigned short f2bf(float f) {
  unsigned int u = __float_as_uint(f);
  u += 0x7FFFu + ((u >> 16) & 1u);           // round-to-nearest-even
  return (unsigned short)(u >> 16);
}

__device__ __forceinline__ v8f wmma_bf16(v16bf a, v16bf b, v8f c) {
  return __builtin_amdgcn_wmma_f32_16x16x32_bf16(false, a, false, b, (short)0, c, false, false);
}

// ---------------- small utility kernels ----------------
__global__ void cvt_bf16_kernel(const float* __restrict__ in, unsigned short* __restrict__ out, size_t n) {
  size_t i = (size_t)blockIdx.x * 256 + threadIdx.x;
  if (i < n) out[i] = f2bf(in[i]);
}

// f32 [K,N] -> bf16 [N,K]  (32x32 tiles through LDS; both sides coalesced)
__global__ __launch_bounds__(256)
void cvtT_bf16_kernel(const float* __restrict__ in, unsigned short* __restrict__ out, int K, int N) {
  __shared__ unsigned short t[32][33];
  int n0 = blockIdx.x * 32, k0 = blockIdx.y * 32;
  int tx = threadIdx.x & 31, ty = threadIdx.x >> 5;
#pragma unroll
  for (int j = 0; j < 4; j++)
    t[ty + j * 8][tx] = f2bf(in[(size_t)(k0 + ty + j * 8) * N + n0 + tx]);
  __syncthreads();
#pragma unroll
  for (int j = 0; j < 4; j++)
    out[(size_t)(n0 + ty + j * 8) * K + k0 + tx] = t[tx][ty + j * 8];
}

__global__ void embed_kernel(const int* __restrict__ ids, const float* __restrict__ tok,
                             const float* __restrict__ pos, float* __restrict__ x, int total) {
  int i = blockIdx.x * 256 + threadIdx.x;
  if (i >= total) return;
  int d  = i % Dd;
  int s  = (i / Dd) % Ss;
  int bs = i / Dd;                       // b*S + s
  x[i] = tok[(size_t)ids[bs] * Dd + d] + pos[(size_t)s * Dd + d];
}

// one block (256 threads) per row of length 1024; writes bf16 (GEMM A input)
__global__ __launch_bounds__(256)
void layernorm_kernel(const float* __restrict__ x, const float* __restrict__ w,
                      const float* __restrict__ b, unsigned short* __restrict__ outb) {
  __shared__ float sbuf[256], s2buf[256];
  int row = blockIdx.x, t = threadIdx.x;
  const float* xr = x + (size_t)row * Dd;
  float v4[4], s = 0.f, s2 = 0.f;
#pragma unroll
  for (int i = 0; i < 4; i++) { float f = xr[t + i * 256]; v4[i] = f; s += f; s2 += f * f; }
  sbuf[t] = s; s2buf[t] = s2; __syncthreads();
  for (int off = 128; off > 0; off >>= 1) {
    if (t < off) { sbuf[t] += sbuf[t + off]; s2buf[t] += s2buf[t + off]; }
    __syncthreads();
  }
  float mean = sbuf[0] * (1.0f / Dd);
  float var  = s2buf[0] * (1.0f / Dd) - mean * mean;
  float inv  = rsqrtf(var + 1e-5f);
#pragma unroll
  for (int i = 0; i < 4; i++) {
    int c = t + i * 256;
    outb[(size_t)row * Dd + c] = f2bf((v4[i] - mean) * inv * w[c] + b[c]);
  }
}

// qkv f32 [B,S,3,H,DH] -> q/k/v bf16 [B,H,S,DH]
__global__ void split_qkv_kernel(const float* __restrict__ qkv, unsigned short* __restrict__ qb,
                                 unsigned short* __restrict__ kb, unsigned short* __restrict__ vb,
                                 int total) {
  int i = blockIdx.x * 256 + threadIdx.x;   // over B*S*D
  if (i >= total) return;
  int dh = i % DHh;
  int h  = (i / DHh) % Hh;
  int s  = (i / Dd) % Ss;
  int b  = i / (Dd * Ss);
  size_t src = ((size_t)(b * Ss + s)) * 3 * Dd + h * DHh + dh;
  size_t dst = (((size_t)(b * Hh + h)) * Ss + s) * DHh + dh;
  qb[dst] = f2bf(qkv[src]);
  kb[dst] = f2bf(qkv[src + Dd]);
  vb[dst] = f2bf(qkv[src + 2 * Dd]);
}

// ---------------- tiled bf16 WMMA GEMM (double-buffered LDS) ----------------
// C[M,N] = act( A[M,K](bf16) @ B[N,K](bf16) + bias ) (+res)
// Block tile 128x128, 8 waves -> each wave 32x64 (2x4 WMMA tiles), K-step 32.
enum { FB_BIAS = 2, FB_RES = 4, FB_GELU = 8, FB_OUTF = 16, FB_OUTB = 32 };

template <int FLAGS>
__global__ __launch_bounds__(256)
void gemm_bf16_kernel(const unsigned short* __restrict__ A, const unsigned short* __restrict__ B,
                      const float* __restrict__ bias, const float* __restrict__ res,
                      float* __restrict__ Cf, unsigned short* __restrict__ Cb,
                      int M, int N, int K) {
  __shared__ unsigned short As[2][128 * 32];   // As[buf][m][k]
  __shared__ unsigned short Bs[2][128 * 32];   // Bs[buf][n][k]
  const int tid = threadIdx.x, lane = tid & 31, wave = tid >> 5;
  const int half = lane >> 4, r16 = lane & 15;
  const int wm = wave >> 1, wn = wave & 1;
  const int row0 = blockIdx.x * 128, col0 = blockIdx.y * 128;

  // loader: each thread owns 16 contiguous K-elems of one A row and one B row
  const int lr = tid >> 1, lc = (tid & 1) * 16;
  const unsigned short* ga = A + (size_t)(row0 + lr) * K + lc;
  const unsigned short* gb = B + (size_t)(col0 + lr) * K + lc;
  unsigned short* la = &As[0][lr * 32 + lc];
  unsigned short* lb = &Bs[0][lr * 32 + lc];
  const int bufStride = 128 * 32;
  const int nt = K / 32;

  v8f acc[2][4] = {};

#if USE_ASYNC_LDS
  ASYNC_CP16(ga, la); ASYNC_CP16(ga + 8, la + 8);
  ASYNC_CP16(gb, lb); ASYNC_CP16(gb + 8, lb + 8);
  __builtin_amdgcn_s_wait_asynccnt(0);
  __syncthreads();
#else
  {
    uint4 a0 = *(const uint4*)(ga), a1 = *(const uint4*)(ga + 8);
    uint4 b0 = *(const uint4*)(gb), b1 = *(const uint4*)(gb + 8);
    *(uint4*)(la) = a0; *(uint4*)(la + 8) = a1;
    *(uint4*)(lb) = b0; *(uint4*)(lb + 8) = b1;
  }
  __syncthreads();
#endif

  int buf = 0;
  for (int it = 0; it < nt; ++it) {
    const int nbuf = buf ^ 1;
    const bool more = (it + 1 < nt);
#if USE_ASYNC_LDS
    if (more) {   // fire next tile's async copies; they land while we compute
      const unsigned short* gan = ga + (it + 1) * 32;
      const unsigned short* gbn = gb + (it + 1) * 32;
      unsigned short* lan = la + nbuf * bufStride;
      unsigned short* lbn = lb + nbuf * bufStride;
      ASYNC_CP16(gan, lan); ASYNC_CP16(gan + 8, lan + 8);
      ASYNC_CP16(gbn, lbn); ASYNC_CP16(gbn + 8, lbn + 8);
    }
#else
    uint4 pa0, pa1, pb0, pb1;
    if (more) {   // global -> regs now; wait happens after the WMMAs at store time
      const unsigned short* gan = ga + (it + 1) * 32;
      const unsigned short* gbn = gb + (it + 1) * 32;
      pa0 = *(const uint4*)(gan); pa1 = *(const uint4*)(gan + 8);
      pb0 = *(const uint4*)(gbn); pb1 = *(const uint4*)(gbn + 8);
    }
#endif

    // fragments + WMMAs on current buffer
    Frag af[2], bfr[4];
#pragma unroll
    for (int mi = 0; mi < 2; mi++) {    // A 16x32: lanes<16 K0-15; lanes>=16 K16-31
      const unsigned short* ap = &As[buf][(wm * 32 + mi * 16 + r16) * 32] + half * 16;
      af[mi].q[0] = *(const uint4*)(ap);
      af[mi].q[1] = *(const uint4*)(ap + 8);
    }
#pragma unroll
    for (int ni = 0; ni < 4; ni++) {    // B 32x16: lanes<16 K0-15; lanes>=16 K16-31
      const unsigned short* bp = &Bs[buf][(wn * 64 + ni * 16 + r16) * 32] + half * 16;
      bfr[ni].q[0] = *(const uint4*)(bp);
      bfr[ni].q[1] = *(const uint4*)(bp + 8);
    }
#pragma unroll
    for (int mi = 0; mi < 2; mi++)
#pragma unroll
      for (int ni = 0; ni < 4; ni++)
        acc[mi][ni] = wmma_bf16(af[mi].v, bfr[ni].v, acc[mi][ni]);

#if USE_ASYNC_LDS
    __builtin_amdgcn_s_wait_asynccnt(0);
#else
    if (more) {
      unsigned short* lan = la + nbuf * bufStride;
      unsigned short* lbn = lb + nbuf * bufStride;
      *(uint4*)(lan) = pa0; *(uint4*)(lan + 8) = pa1;
      *(uint4*)(lbn) = pb0; *(uint4*)(lbn + 8) = pb1;
    }
#endif
    __syncthreads();
    buf = nbuf;
  }

  // epilogue (C layout: VGPR v -> row v + 8*half, col = lane&15)
#pragma unroll
  for (int mi = 0; mi < 2; mi++)
#pragma unroll
    for (int ni = 0; ni < 4; ni++) {
      int gc = col0 + wn * 64 + ni * 16 + r16;
      float bval = 0.f;
      if constexpr (FLAGS & FB_BIAS) bval = bias[gc];
#pragma unroll
      for (int v = 0; v < 8; v++) {
        int gr = row0 + wm * 32 + mi * 16 + half * 8 + v;
        size_t idx = (size_t)gr * N + gc;
        float val = acc[mi][ni][v] + bval;
        if constexpr (FLAGS & FB_RES)  val += res[idx];
        if constexpr (FLAGS & FB_GELU) val = 0.5f * val * (1.f + erff(val * 0.70710678118654752f));
        if constexpr (FLAGS & FB_OUTF) Cf[idx] = val;
        if constexpr (FLAGS & FB_OUTB) Cb[idx] = f2bf(val);
      }
    }
}

// ---------------- flash attention (bf16 WMMA, f32 online softmax) ----------------
// grid: (S/128, H, B); 8 waves, each owns 16 query rows; kv blocks of 32.
__global__ __launch_bounds__(256)
void attn_kernel(const unsigned short* __restrict__ Q, const unsigned short* __restrict__ K,
                 const unsigned short* __restrict__ V, unsigned short* __restrict__ Obf,
                 float scale) {
  __shared__ unsigned short Vs[64 * 32];      // Vs[dh][kv]
  __shared__ unsigned short Ps[8][16 * 32];   // per-wave P tile [m][kv]
  const int tid = threadIdx.x, lane = tid & 31, wave = tid >> 5;
  const int half = lane >> 4, r16 = lane & 15;
  const int b = blockIdx.z, h = blockIdx.y, qt = blockIdx.x;
  const size_t base = ((size_t)(b * Hh + h)) * Ss * DHh;
  const unsigned short* Qp = Q + base;
  const unsigned short* Kp = K + base;
  const unsigned short* Vp = V + base;
  const int q0 = qt * 128 + wave * 16;

  // Q fragments: a0 covers dh 0..31, a1 covers dh 32..63
  Frag a0, a1;
  {
    const unsigned short* qrow = Qp + (size_t)(q0 + r16) * DHh;
    int c = half * 16;
    a0.q[0] = *(const uint4*)(qrow + c);
    a0.q[1] = *(const uint4*)(qrow + c + 8);
    a1.q[0] = *(const uint4*)(qrow + 32 + c);
    a1.q[1] = *(const uint4*)(qrow + 32 + c + 8);
  }

  const float NEG = -1e30f;
  float m_run[8], l_run[8];
#pragma unroll
  for (int v = 0; v < 8; v++) { m_run[v] = NEG; l_run[v] = 0.f; }
  v8f acc[4] = {};
  const int qmax = q0 + 15;
  const int kv_end = qt * 128 + 128;

  for (int kvb = 0; kvb < kv_end; kvb += 32) {
    { // cooperative transposed V stage: V[kvb+kv][dh] -> Vs[dh][kv]
      int kv = tid >> 3, dh0 = (tid & 7) * 8;
      const unsigned short* src = Vp + (size_t)(kvb + kv) * DHh + dh0;
#pragma unroll
      for (int i = 0; i < 8; i++) Vs[(dh0 + i) * 32 + kv] = src[i];
    }
    __syncthreads();

    if (kvb <= qmax) {
      const bool needMask = (kvb + 31) > q0;   // diagonal block for this wave?
      // scores: two 16x16 tiles (cols kvb+0..15 / kvb+16..31), K=64 => 2 WMMA each.
      const unsigned short* k0row = Kp + (size_t)(kvb + r16) * DHh;
      const unsigned short* k1row = Kp + (size_t)(kvb + 16 + r16) * DHh;
      Frag b0lo, b0hi, b1lo, b1hi;
      b0lo.q[0] = *(const uint4*)(k0row + half * 16);
      b0lo.q[1] = *(const uint4*)(k0row + half * 16 + 8);
      b0hi.q[0] = *(const uint4*)(k0row + 32 + half * 16);
      b0hi.q[1] = *(const uint4*)(k0row + 32 + half * 16 + 8);
      b1lo.q[0] = *(const uint4*)(k1row + half * 16);
      b1lo.q[1] = *(const uint4*)(k1row + half * 16 + 8);
      b1hi.q[0] = *(const uint4*)(k1row + 32 + half * 16);
      b1hi.q[1] = *(const uint4*)(k1row + 32 + half * 16 + 8);

      v8f c0 = {}, c1 = {};
      c0 = wmma_bf16(a0.v, b0lo.v, c0); c0 = wmma_bf16(a1.v, b0hi.v, c0);
      c1 = wmma_bf16(a0.v, b1lo.v, c1); c1 = wmma_bf16(a1.v, b1hi.v, c1);

      float p0[8], p1[8];
#pragma unroll
      for (int v = 0; v < 8; v++) {
        int q = q0 + half * 8 + v;
        float s0 = c0[v] * scale;
        float s1 = c1[v] * scale;
        if (needMask) {
          if (kvb + r16 > q)      s0 = NEG;
          if (kvb + 16 + r16 > q) s1 = NEG;
        }
        float rm = fmaxf(s0, s1);
        rm = fmaxf(rm, __shfl_xor(rm, 1, 32));
        rm = fmaxf(rm, __shfl_xor(rm, 2, 32));
        rm = fmaxf(rm, __shfl_xor(rm, 4, 32));
        rm = fmaxf(rm, __shfl_xor(rm, 8, 32));
        float mnew = fmaxf(m_run[v], rm);
        float corr = __expf(m_run[v] - mnew);
        float e0 = __expf(s0 - mnew), e1 = __expf(s1 - mnew);
        float rs = e0 + e1;
        rs += __shfl_xor(rs, 1, 32);
        rs += __shfl_xor(rs, 2, 32);
        rs += __shfl_xor(rs, 4, 32);
        rs += __shfl_xor(rs, 8, 32);
        l_run[v] = l_run[v] * corr + rs;
        m_run[v] = mnew;
#pragma unroll
        for (int n = 0; n < 4; n++) acc[n][v] *= corr;
        p0[v] = e0; p1[v] = e1;
      }

      // C-layout -> A-layout transpose of P through per-wave LDS
      unsigned short* ps = Ps[wave];
#pragma unroll
      for (int v = 0; v < 8; v++) {
        int m = half * 8 + v;
        ps[m * 32 + r16]      = f2bf(p0[v]);
        ps[m * 32 + 16 + r16] = f2bf(p1[v]);
      }
      Frag pa;
      {
        const unsigned short* pr = ps + r16 * 32 + half * 16;
        pa.q[0] = *(const uint4*)(pr);
        pa.q[1] = *(const uint4*)(pr + 8);
      }
      // O += P[16x32] @ V[32x64]
#pragma unroll
      for (int n = 0; n < 4; n++) {
        Frag bv;
        const unsigned short* vr = &Vs[(n * 16 + r16) * 32] + half * 16;
        bv.q[0] = *(const uint4*)(vr);
        bv.q[1] = *(const uint4*)(vr + 8);
        acc[n] = wmma_bf16(pa.v, bv.v, acc[n]);
      }
    }
    __syncthreads();
  }

  // O = acc / l  -> bf16 [B*S, D] at col h*64+dh
#pragma unroll
  for (int v = 0; v < 8; v++) {
    int q = q0 + half * 8 + v;
    float invl = 1.0f / l_run[v];
#pragma unroll
    for (int n = 0; n < 4; n++) {
      size_t idx = ((size_t)b * Ss + q) * Dd + h * DHh + n * 16 + r16;
      Obf[idx] = f2bf(acc[n][v] * invl);
    }
  }
}

// ---------------- host launcher ----------------
extern "C" void kernel_launch(void* const* d_in, const int* in_sizes, int n_in,
                              void* d_out, int out_size, void* d_ws, size_t ws_size,
                              hipStream_t stream) {
  (void)in_sizes; (void)n_in; (void)out_size; (void)ws_size;
  const int*   ids     = (const int*)d_in[0];
  const float* tok_emb = (const float*)d_in[1];
  const float* pos_emb = (const float*)d_in[2];
  const float* ln1_w   = (const float*)d_in[3];
  const float* ln1_b   = (const float*)d_in[4];
  const float* qkv_w   = (const float*)d_in[5];
  const float* qkv_b   = (const float*)d_in[6];
  const float* out_w   = (const float*)d_in[7];
  const float* out_b   = (const float*)d_in[8];
  const float* ln2_w   = (const float*)d_in[9];
  const float* ln2_b   = (const float*)d_in[10];
  const float* fc1_w   = (const float*)d_in[11];
  const float* fc1_b   = (const float*)d_in[12];
  const float* fc2_w   = (const float*)d_in[13];
  const float* fc2_b   = (const float*)d_in[14];
  const float* lnf_w   = (const float*)d_in[15];
  const float* lnf_b   = (const float*)d_in[16];
  float* logits = (float*)d_out;

  char* p = (char*)d_ws;
  auto carve = [&](size_t bytes) -> void* {
    void* r = (void*)p; p += (bytes + 255) & ~(size_t)255; return r;
  };
  const size_t BS = (size_t)Bb * Ss;   // 2048
  float*          x    = (float*)carve(BS * Dd * 4);
  float*          qkvf = (float*)carve(BS * 3 * Dd * 4);
  unsigned short* qb   = (unsigned short*)carve(BS * Dd * 2);
  unsigned short* kb   = (unsigned short*)carve(BS * Dd * 2);
  unsigned short* vb   = (unsigned short*)carve(BS * Dd * 2);
  unsigned short* actb = (unsigned short*)carve(BS * Dd * 2);
  unsigned short* midb = (unsigned short*)carve(BS * Ff * 2);
  unsigned short* wb   = (unsigned short*)carve((size_t)Dd * Ff * 2);
  unsigned short* temb = (unsigned short*)carve((size_t)Vv * Dd * 2);

  auto cvt = [&](const float* src, unsigned short* dst, size_t n) {
    cvt_bf16_kernel<<<(unsigned)((n + 255) / 256), 256, 0, stream>>>(src, dst, n);
  };
  auto cvtT = [&](const float* src, unsigned short* dst, int K, int N) {   // [K,N] -> [N,K] bf16
    cvtT_bf16_kernel<<<dim3(N / 32, K / 32), 256, 0, stream>>>(src, dst, K, N);
  };

  // token embedding table -> bf16 (already [V,K] = [N,K] for the logits GEMM)
  cvt(tok_emb, temb, (size_t)Vv * Dd);

  { // x = tok_emb[ids] + pos_emb
    int total = (int)(BS * Dd);
    embed_kernel<<<(total + 255) / 256, 256, 0, stream>>>(ids, tok_emb, pos_emb, x, total);
  }

  for (int l = 0; l < Ll; l++) {
    // ---- attention block ----
    layernorm_kernel<<<(unsigned)BS, 256, 0, stream>>>(x, ln1_w + (size_t)l * Dd, ln1_b + (size_t)l * Dd, actb);
    cvtT(qkv_w + (size_t)l * Dd * 3 * Dd, wb, Dd, 3 * Dd);
    gemm_bf16_kernel<FB_BIAS | FB_OUTF><<<dim3(BS / 128, (3 * Dd) / 128), 256, 0, stream>>>(
        actb, wb, qkv_b + (size_t)l * 3 * Dd, nullptr, qkvf, nullptr, (int)BS, 3 * Dd, Dd);
    { int total = (int)(BS * Dd);
      split_qkv_kernel<<<(total + 255) / 256, 256, 0, stream>>>(qkvf, qb, kb, vb, total); }
    attn_kernel<<<dim3(Ss / 128, Hh, Bb), 256, 0, stream>>>(qb, kb, vb, actb, 0.125f);
    cvtT(out_w + (size_t)l * Dd * Dd, wb, Dd, Dd);
    gemm_bf16_kernel<FB_BIAS | FB_RES | FB_OUTF><<<dim3(BS / 128, Dd / 128), 256, 0, stream>>>(
        actb, wb, out_b + (size_t)l * Dd, x, x, nullptr, (int)BS, Dd, Dd);

    // ---- MLP block ----
    layernorm_kernel<<<(unsigned)BS, 256, 0, stream>>>(x, ln2_w + (size_t)l * Dd, ln2_b + (size_t)l * Dd, actb);
    cvtT(fc1_w + (size_t)l * Dd * Ff, wb, Dd, Ff);
    gemm_bf16_kernel<FB_BIAS | FB_GELU | FB_OUTB><<<dim3(BS / 128, Ff / 128), 256, 0, stream>>>(
        actb, wb, fc1_b + (size_t)l * Ff, nullptr, nullptr, midb, (int)BS, Ff, Dd);
    cvtT(fc2_w + (size_t)l * Ff * Dd, wb, Ff, Dd);
    gemm_bf16_kernel<FB_BIAS | FB_RES | FB_OUTF><<<dim3(BS / 128, Dd / 128), 256, 0, stream>>>(
        midb, wb, fc2_b + (size_t)l * Dd, x, x, nullptr, (int)BS, Dd, Ff);
  }

  // ---- final LN + logits = x @ tok_emb^T ----
  layernorm_kernel<<<(unsigned)BS, 256, 0, stream>>>(x, lnf_w, lnf_b, actb);
  gemm_bf16_kernel<FB_OUTF><<<dim3(BS / 128, Vv / 128), 256, 0, stream>>>(
      actb, temb, nullptr, nullptr, logits, nullptr, (int)BS, Vv, Dd);
}
